// GlimpseDecoder_79010218377493
// MI455X (gfx1250) — matrix operand order repaired and checked by reading, not compiled, not run on hardware
//
#include <hip/hip_runtime.h>
#include <hip/hip_bf16.h>

// Problem constants (match reference)
#define B_     64
#define N_     1000
#define D_     128
#define H_     8
#define DH_    16
#define STEPS_ 999
#define NEGV   (-1e9f)

typedef __attribute__((ext_vector_type(16))) __bf16 v16bf;
typedef __attribute__((ext_vector_type(8)))  float  v8f;

__device__ __forceinline__ float bf2f(unsigned int u) {
    union { unsigned int i; float f; } x; x.i = u << 16; return x.f;
}
__device__ __forceinline__ unsigned short f2bf(float f) {
    union { float f; unsigned int i; } x; x.f = f; return (unsigned short)(x.i >> 16);
}

// ---------------------------------------------------------------------------
// Kernel 1: qkv = embeddings @ W_node via v_wmma_f32_16x16x32_bf16.
// Grid: (64000/64 row-blocks, 384/16 col-tiles), 128 threads (4 waves).
// Each wave computes one 16x16 f32 tile with 4 chained K=32 bf16 WMMAs.
// Results are scattered bf16 into per-head gK/gV and row-major lK caches
// (total 48 MB bf16 -> resident in the 192 MB L2 for the decode scan).
// ---------------------------------------------------------------------------
__global__ __launch_bounds__(128)
void qkv_wmma_kernel(const float* __restrict__ emb,   // [64*1000, 128]
                     const float* __restrict__ Wn,    // [128, 384]
                     unsigned short* __restrict__ gK, // [B,H,N,DH] bf16
                     unsigned short* __restrict__ gV, // [B,H,N,DH] bf16
                     unsigned short* __restrict__ lK) // [B,N,D]    bf16
{
    __shared__ unsigned short As[64 * 128];  // 64 rows x K=128, bf16
    __shared__ unsigned short Bs[128 * 16];  // K=128 x 16 cols, bf16

    const int tid     = threadIdx.x;
    const int rowBase = blockIdx.x * 64;
    const int cb      = blockIdx.y * 16;

    // Stage A rows (coalesced) with fp32 -> bf16 truncation
    for (int i = tid; i < 64 * 128; i += 128)
        As[i] = f2bf(emb[(size_t)rowBase * 128 + i]);
    // Stage the 16-column slice of W_node
    for (int i = tid; i < 128 * 16; i += 128) {
        int k = i >> 4, j = i & 15;
        Bs[i] = f2bf(Wn[k * 384 + cb + j]);
    }
    __syncthreads();

    const int wave  = tid >> 5, lane = tid & 31;
    const int m0    = wave * 16;
    const int mrowA = lane & 15;
    const int kselA = (lane < 16) ? 0 : 16;  // 16-bit A 16x32 lane split
    const int ncol  = lane & 15;
    const int kselB = (lane < 16) ? 0 : 16;  // 16-bit B 32x16 lane split

    v8f acc = {0.f, 0.f, 0.f, 0.f, 0.f, 0.f, 0.f, 0.f};

#pragma unroll
    for (int kk = 0; kk < 4; ++kk) {
        const int kb = kk * 32;
        union { v16bf v; unsigned short u[16]; } a, b;
#pragma unroll
        for (int e = 0; e < 16; ++e) {
            // A fragment: 16 consecutive K per half-wave (matches B)
            int kA = kselA + e;
            a.u[e] = As[(m0 + mrowA) * 128 + kb + kA];
            // B fragment: lane = column, 16 consecutive K per half-wave
            b.u[e] = Bs[(kb + kselB + e) * 16 + ncol];
        }
        acc = __builtin_amdgcn_wmma_f32_16x16x32_bf16(
            false, a.v, false, b.v, (short)0, acc, false, false);
    }

    // C layout: VGPR i -> M = i + 8*(lane/16); N = lane%16
    const int j = cb + ncol;
#pragma unroll
    for (int i = 0; i < 8; ++i) {
        int mrow = i + ((lane >> 4) << 3);
        int grow = rowBase + m0 + mrow;          // global row in [0, 64000)
        int bb   = grow / N_, n = grow % N_;
        unsigned short v = f2bf(acc[i]);
        if (j < 128) {
            int h = j >> 4, d = j & 15;
            gK[(((size_t)bb * H_ + h) * N_ + n) * DH_ + d] = v;
        } else if (j < 256) {
            int jj = j - 128, h = jj >> 4, d = jj & 15;
            gV[(((size_t)bb * H_ + h) * N_ + n) * DH_ + d] = v;
        } else {
            lK[((size_t)bb * N_ + n) * D_ + (j - 256)] = v;
        }
    }
}

// ---------------------------------------------------------------------------
// Kernel 2: persistent per-batch decoder. One block per b (grid=64),
// 512 threads = 16 wave32s. Runs all 999 greedy steps with LDS-resident
// mask/tour/compat/logits; K/V/lK stream from the L2-resident bf16 cache.
// Glimpse accumulation uses LDS float atomics (ds_add_f32).
// ---------------------------------------------------------------------------
__global__ __launch_bounds__(512)
void decoder_kernel(const float* __restrict__ nodes, // [B,N,2]
                    const float* __restrict__ emb,   // [B,N,D]
                    const float* __restrict__ Wstep, // [D,D]
                    const unsigned short* __restrict__ gK,
                    const unsigned short* __restrict__ gV,
                    const unsigned short* __restrict__ lK,
                    float* __restrict__ out_logp,    // [B,999,1000]
                    float* __restrict__ out_pi,      // [B,999] (as float)
                    float* __restrict__ out_cost)    // [B]
{
    __shared__ float sCompat[H_][N_];   // 32 KB
    __shared__ float sLogit[N_];        // 4 KB
    __shared__ float sE[D_], sQ[D_], sGl[D_];
    __shared__ float sRed[512];
    __shared__ int   sRedI[512];
    __shared__ int   sTour[N_];
    __shared__ unsigned char sMask[N_];
    __shared__ int   sCur;

    const int b   = blockIdx.x;
    const int tid = threadIdx.x;
    const int h   = tid >> 6;   // 64-thread group per head
    const int g   = tid & 63;
    const float scale_g = 0.25f;                  // 1/sqrt(16)
    const float scale_l = 0.08838834764831845f;   // 1/sqrt(128)

    for (int n = tid; n < N_; n += 512) sMask[n] = 0;
    if (tid == 0) { sMask[0] = 1; sCur = 0; sTour[0] = 0; }
    __syncthreads();

    const unsigned short* gKb = gK + (size_t)b * H_ * N_ * DH_;
    const unsigned short* gVb = gV + (size_t)b * H_ * N_ * DH_;
    const unsigned short* lKb = lK + (size_t)b * N_ * D_;

    for (int t = 0; t < STEPS_; ++t) {
        // ---- step context + q = e @ W_step -----------------------------
        if (tid < D_)            sE[tid] = emb[((size_t)b * N_ + sCur) * D_ + tid];
        else if (tid < 2 * D_)   sGl[tid - D_] = 0.f;
        __syncthreads();
        if (tid < D_) {
            float q = 0.f;
            for (int d = 0; d < D_; ++d)
                q = fmaf(sE[d], Wstep[d * D_ + tid], q);
            sQ[tid] = q;
        }
        __syncthreads();

        // ---- compat[h][n] = scale_g * q_h . K_h[n], masked -------------
        float qh[DH_];
#pragma unroll
        for (int d = 0; d < DH_; ++d) qh[d] = sQ[h * DH_ + d];

        float mx = NEGV;
        for (int n = g; n < N_; n += 64) {
            const uint4* p = reinterpret_cast<const uint4*>(
                gKb + ((size_t)h * N_ + n) * DH_);
            uint4 x0 = p[0], x1 = p[1];
            unsigned int w[8] = {x0.x, x0.y, x0.z, x0.w, x1.x, x1.y, x1.z, x1.w};
            float c = 0.f;
#pragma unroll
            for (int i = 0; i < 8; ++i) {
                c = fmaf(qh[2 * i],     bf2f(w[i] & 0xffffu), c);
                c = fmaf(qh[2 * i + 1], bf2f(w[i] >> 16),     c);
            }
            c *= scale_g;
            if (sMask[n]) c = NEGV;
            sCompat[h][n] = c;
            mx = fmaxf(mx, c);
        }
        sRed[tid] = mx;
        __syncthreads();
        for (int off = 32; off >= 1; off >>= 1) {
            if (g < off) sRed[tid] = fmaxf(sRed[tid], sRed[tid + off]);
            __syncthreads();
        }
        const float m = sRed[h << 6];
        __syncthreads();

        float sm = 0.f;
        for (int n = g; n < N_; n += 64) sm += __expf(sCompat[h][n] - m);
        sRed[tid] = sm;
        __syncthreads();
        for (int off = 32; off >= 1; off >>= 1) {
            if (g < off) sRed[tid] += sRed[tid + off];
            __syncthreads();
        }
        const float invs = 1.f / sRed[h << 6];
        __syncthreads();

        // ---- glimpse_h = softmax . V_h  (LDS float atomics) ------------
        float accv[DH_];
#pragma unroll
        for (int d = 0; d < DH_; ++d) accv[d] = 0.f;
        for (int n = g; n < N_; n += 64) {
            float wgt = __expf(sCompat[h][n] - m) * invs;
            const uint4* p = reinterpret_cast<const uint4*>(
                gVb + ((size_t)h * N_ + n) * DH_);
            uint4 x0 = p[0], x1 = p[1];
            unsigned int w[8] = {x0.x, x0.y, x0.z, x0.w, x1.x, x1.y, x1.z, x1.w};
#pragma unroll
            for (int i = 0; i < 8; ++i) {
                accv[2 * i]     = fmaf(wgt, bf2f(w[i] & 0xffffu), accv[2 * i]);
                accv[2 * i + 1] = fmaf(wgt, bf2f(w[i] >> 16),     accv[2 * i + 1]);
            }
        }
#pragma unroll
        for (int d = 0; d < DH_; ++d) atomicAdd(&sGl[h * DH_ + d], accv[d]);
        __syncthreads();

        // ---- logits[n] = 10*tanh(scale_l * glimpse . lK[n]), masked ----
        float lmax = NEGV; int larg = 0;
        for (int n = tid; n < N_; n += 512) {
            if (n + 512 < N_)
                __builtin_prefetch(lKb + (size_t)(n + 512) * D_, 0, 0);
            const uint4* p = reinterpret_cast<const uint4*>(lKb + (size_t)n * D_);
            float c = 0.f;
#pragma unroll
            for (int q8 = 0; q8 < 16; ++q8) {
                uint4 x = p[q8];
                int d0 = q8 * 8;
                c = fmaf(sGl[d0 + 0], bf2f(x.x & 0xffffu), c);
                c = fmaf(sGl[d0 + 1], bf2f(x.x >> 16),     c);
                c = fmaf(sGl[d0 + 2], bf2f(x.y & 0xffffu), c);
                c = fmaf(sGl[d0 + 3], bf2f(x.y >> 16),     c);
                c = fmaf(sGl[d0 + 4], bf2f(x.z & 0xffffu), c);
                c = fmaf(sGl[d0 + 5], bf2f(x.z >> 16),     c);
                c = fmaf(sGl[d0 + 6], bf2f(x.w & 0xffffu), c);
                c = fmaf(sGl[d0 + 7], bf2f(x.w >> 16),     c);
            }
            c = 10.f * tanhf(c * scale_l);
            if (sMask[n]) c = NEGV;
            sLogit[n] = c;
            if (c > lmax) { lmax = c; larg = n; }
        }
        sRed[tid] = lmax; sRedI[tid] = larg;
        __syncthreads();
        for (int off = 256; off >= 1; off >>= 1) {
            if (tid < off) {
                float v2 = sRed[tid + off]; int i2 = sRedI[tid + off];
                if (v2 > sRed[tid] || (v2 == sRed[tid] && i2 < sRedI[tid])) {
                    sRed[tid] = v2; sRedI[tid] = i2;
                }
            }
            __syncthreads();
        }
        const float m2 = sRed[0];
        const int   sel = sRedI[0];
        __syncthreads();

        float psum = 0.f;
        for (int n = tid; n < N_; n += 512) psum += __expf(sLogit[n] - m2);
        sRed[tid] = psum;
        __syncthreads();
        for (int off = 256; off >= 1; off >>= 1) {
            if (tid < off) sRed[tid] += sRed[tid + off];
            __syncthreads();
        }
        const float logZ = m2 + __logf(sRed[0]);

        float* lp = out_logp + ((size_t)b * STEPS_ + t) * N_;
        for (int n = tid; n < N_; n += 512) lp[n] = sLogit[n] - logZ;

        if (tid == 0) {
            sMask[sel] = 1; sCur = sel; sTour[t + 1] = sel;
            out_pi[(size_t)b * STEPS_ + t] = (float)sel;
        }
        __syncthreads();
    }

    // ---- closed tour cost ---------------------------------------------
    float csum = 0.f;
    for (int i = tid; i < N_; i += 512) {
        int a = sTour[i], c = sTour[(i + 1) % N_];
        float ax = nodes[((size_t)b * N_ + a) * 2 + 0];
        float ay = nodes[((size_t)b * N_ + a) * 2 + 1];
        float cx = nodes[((size_t)b * N_ + c) * 2 + 0];
        float cy = nodes[((size_t)b * N_ + c) * 2 + 1];
        float dx = ax - cx, dy = ay - cy;
        csum += sqrtf(dx * dx + dy * dy);
    }
    sRed[tid] = csum;
    __syncthreads();
    for (int off = 256; off >= 1; off >>= 1) {
        if (tid < off) sRed[tid] += sRed[tid + off];
        __syncthreads();
    }
    if (tid == 0) out_cost[b] = sRed[0];
}

// ---------------------------------------------------------------------------
extern "C" void kernel_launch(void* const* d_in, const int* in_sizes, int n_in,
                              void* d_out, int out_size, void* d_ws, size_t ws_size,
                              hipStream_t stream) {
    (void)in_sizes; (void)n_in; (void)out_size; (void)ws_size;
    const float* nodes = (const float*)d_in[0];
    const float* emb   = (const float*)d_in[1];
    const float* Wn    = (const float*)d_in[2];
    // d_in[3] = W_fixed: feeds only graph_context, which never reaches an output.
    const float* Wstep = (const float*)d_in[4];

    // Workspace: bf16 caches gK | gV | lK  (16 MB each, 48 MB total -> L2 resident)
    unsigned short* gK = (unsigned short*)d_ws;
    unsigned short* gV = gK + (size_t)B_ * H_ * N_ * DH_;
    unsigned short* lK = gV + (size_t)B_ * H_ * N_ * DH_;

    float* out      = (float*)d_out;
    float* out_logp = out;                                  // [B,999,1000]
    float* out_pi   = out + (size_t)B_ * STEPS_ * N_;       // [B,999]
    float* out_cost = out_pi + (size_t)B_ * STEPS_;         // [B]

    dim3 g1(1000, 24);
    qkv_wmma_kernel<<<g1, dim3(128), 0, stream>>>(emb, Wn, gK, gV, lK);
    decoder_kernel<<<dim3(64), dim3(512), 0, stream>>>(
        nodes, emb, Wstep, gK, gV, lK, out_logp, out_pi, out_cost);
}
